// CRFTagger_7284264534265
// MI455X (gfx1250) — matrix operand order repaired and checked by reading, not compiled, not run on hardware
//
#include <hip/hip_runtime.h>
#include <hip/hip_bf16.h>
#include <math.h>

// Problem constants (from reference)
#define BB   512
#define SS   512
#define WW   3
#define EE   128
#define HH   100     // real hidden dim
#define HP   128     // padded hidden dim for WMMA
#define TT   64
#define NROW (BB*SS) // 262144 flattened (b,s) rows

typedef __attribute__((ext_vector_type(16))) _Float16 v16h;
typedef __attribute__((ext_vector_type(8)))  _Float16 v8h;
typedef __attribute__((ext_vector_type(8)))  float    v8f;

// gfx1250 has a native V_TANH_F32; use it if the builtin exists.
#if __has_builtin(__builtin_amdgcn_tanhf)
  #define FAST_TANH(x) __builtin_amdgcn_tanhf(x)
#elif __has_builtin(__builtin_amdgcn_tanh_f32)
  #define FAST_TANH(x) __builtin_amdgcn_tanh_f32(x)
#else
  #define FAST_TANH(x) tanhf(x)
#endif

// ---------------------------------------------------------------------------
// Kernel 1: fused embedding gather + tanh-MLP via WMMA.
//   Each block = 2 waves; each wave owns a 16-row M-tile of the 262144 rows.
//   A-fragments built in registers straight from the gather (no LDS staging);
//   W1/W2 pre-swizzled in LDS to the per-lane fragment layout so every
//   B-fragment is two contiguous ds_load_b128.
// ---------------------------------------------------------------------------
__global__ __launch_bounds__(64)
void crf_mlp_wmma_kernel(const int*   __restrict__ inputs,     // (NROW,3)
                         const float* __restrict__ emb_table,  // (V,128)
                         const float* __restrict__ W1,         // (128,100)
                         const float* __restrict__ b1,         // (100)
                         const float* __restrict__ W2,         // (100,64)
                         const float* __restrict__ b2,         // (64)
                         float*       __restrict__ em,         // (NROW,64)
                         float*       __restrict__ maskf)      // (NROW)
{
    // B-fragment-swizzled weights: [n_tile][k_step][lane][16 halves]
    __shared__ __attribute__((aligned(32))) _Float16 sW1f[8][4][32][16]; // 32 KB
    __shared__ __attribute__((aligned(32))) _Float16 sW2f[4][4][32][16]; // 16 KB
    __shared__ __attribute__((aligned(16))) _Float16 sH[2][16][HP];      //  8 KB

    const int tid  = threadIdx.x;   // 0..63
    const int wave = tid >> 5;      // 0..1
    const int lane = tid & 31;
    const int mrow = lane & 15;     // A-row / B-col / D-col index for this lane
    const int hi   = lane >> 4;     // lane half-select

    // --- W1 -> LDS, swizzled to fragment order (contiguous LDS stores) ---
    // layout idx = ((nt*4 + kk)*32 + ln)*16 + p ;  n = nt*16+(ln&15),
// k = kk*32 + (p>>3)*16 + (ln>>4)*8 + (p&7)   (B frag: table order, matches A)
    for (int idx = tid; idx < 8 * 4 * 32 * 16; idx += 64) {
        const int p  = idx & 15;
        const int ln = (idx >> 4) & 31;
        const int kk = (idx >> 9) & 3;
        const int nt = idx >> 11;
        const int n  = nt * 16 + (ln & 15);
        const int k  = kk * 32 + (p >> 3) * 16 + (ln >> 4) * 8 + (p & 7);
        const float v = (n < HH) ? W1[k * HH + n] : 0.0f;
        ((_Float16*)sW1f)[idx] = (_Float16)v;
    }
    // --- W2 -> LDS, same fragment swizzle (pad K 100..127 with zero) ---
    for (int idx = tid; idx < 4 * 4 * 32 * 16; idx += 64) {
        const int p  = idx & 15;
        const int ln = (idx >> 4) & 31;
        const int kk = (idx >> 9) & 3;
        const int nt = idx >> 11;
        const int n  = nt * 16 + (ln & 15);
        const int k  = kk * 32 + (p >> 3) * 16 + (ln >> 4) * 8 + (p & 7);
        const float v = (k < HH) ? W2[k * TT + n] : 0.0f;
        ((_Float16*)sW2f)[idx] = (_Float16)v;
    }

    // --- build GEMM1 A-fragments in registers from the embedding gather ---
    // A layout: lane(mrow,hi), halves p=0..7 -> k = 32*kk + 8*hi + p,
    //                          halves p=8..15 -> k = 32*kk + 16 + 8*hi + (p-8)
    const int r0 = (blockIdx.x * 2 + wave) * 16;
    const int r  = r0 + mrow;
    const int i0 = inputs[r * 3 + 0];
    const int i1 = inputs[r * 3 + 1];
    const int i2 = inputs[r * 3 + 2];
    const float* t0 = emb_table + (size_t)i0 * EE;
    const float* t1 = emb_table + (size_t)i1 * EE;
    const float* t2 = emb_table + (size_t)i2 * EE;

    v16h aF[4];
    #pragma unroll
    for (int kk = 0; kk < 4; ++kk) {
        #pragma unroll
        for (int h2 = 0; h2 < 2; ++h2) {
            const int ks = 32 * kk + 16 * h2 + 8 * hi;   // 32B-aligned run of 8
            const float4 u0 = *(const float4*)(t0 + ks);
            const float4 u1 = *(const float4*)(t1 + ks);
            const float4 u2 = *(const float4*)(t2 + ks);
            const float4 w0 = *(const float4*)(t0 + ks + 4);
            const float4 w1 = *(const float4*)(t1 + ks + 4);
            const float4 w2 = *(const float4*)(t2 + ks + 4);
            aF[kk][8 * h2 + 0] = (_Float16)(u0.x + u1.x + u2.x);
            aF[kk][8 * h2 + 1] = (_Float16)(u0.y + u1.y + u2.y);
            aF[kk][8 * h2 + 2] = (_Float16)(u0.z + u1.z + u2.z);
            aF[kk][8 * h2 + 3] = (_Float16)(u0.w + u1.w + u2.w);
            aF[kk][8 * h2 + 4] = (_Float16)(w0.x + w1.x + w2.x);
            aF[kk][8 * h2 + 5] = (_Float16)(w0.y + w1.y + w2.y);
            aF[kk][8 * h2 + 6] = (_Float16)(w0.z + w1.z + w2.z);
            aF[kk][8 * h2 + 7] = (_Float16)(w0.w + w1.w + w2.w);
        }
    }
    __syncthreads();   // weights swizzled into LDS

    // --- GEMM1: h = tanh(emb @ W1 + b1), N = 128 (8 tiles), K = 128 ---
    #pragma unroll
    for (int nt = 0; nt < 8; ++nt) {
        const int ncol = nt * 16 + mrow;
        v8f acc = {0.f, 0.f, 0.f, 0.f, 0.f, 0.f, 0.f, 0.f};
        #pragma unroll
        for (int kk = 0; kk < 4; ++kk) {
            const v16h bf = *(const v16h*)(&sW1f[nt][kk][lane][0]); // 2x ds_load_b128
            acc = __builtin_amdgcn_wmma_f32_16x16x32_f16(
                      false, aF[kk], false, bf, (short)0, acc, false, false);
        }
        const float b1v = (ncol < HH) ? b1[ncol] : 0.0f;
        #pragma unroll
        for (int v = 0; v < 8; ++v)             // D: m = v + 8*hi, n = ncol
            sH[wave][v + 8 * hi][ncol] = (_Float16)FAST_TANH(acc[v] + b1v);
    }
    __syncthreads();

    // --- GEMM2 A-fragments from sH (two aligned 16B runs per k-step) ---
    v16h ah[4];
    #pragma unroll
    for (int kk = 0; kk < 4; ++kk) {
        const v8h ra = *(const v8h*)(&sH[wave][mrow][32 * kk + 8 * hi]);
        const v8h rb = *(const v8h*)(&sH[wave][mrow][32 * kk + 16 + 8 * hi]);
        #pragma unroll
        for (int q = 0; q < 8; ++q) { ah[kk][q] = ra[q]; ah[kk][8 + q] = rb[q]; }
    }

    // --- GEMM2: em = h @ W2 + b2, N = 64 (4 tiles), K = 128 ---
    #pragma unroll
    for (int nt = 0; nt < 4; ++nt) {
        const int ncol = nt * 16 + mrow;        // 0..63
        v8f acc = {0.f, 0.f, 0.f, 0.f, 0.f, 0.f, 0.f, 0.f};
        #pragma unroll
        for (int kk = 0; kk < 4; ++kk) {
            const v16h bf = *(const v16h*)(&sW2f[nt][kk][lane][0]);
            acc = __builtin_amdgcn_wmma_f32_16x16x32_f16(
                      false, ah[kk], false, bf, (short)0, acc, false, false);
        }
        const float b2v = b2[ncol];
        #pragma unroll
        for (int v = 0; v < 8; ++v)
            em[(size_t)(r0 + v + 8 * hi) * TT + ncol] = acc[v] + b2v;
    }

    // --- mask for the 16 rows of this wave ---
    if (lane < 16) {
        const int rr = r0 + lane;
        const int j0 = inputs[rr * 3 + 0];
        const int j1 = inputs[rr * 3 + 1];
        const int j2 = inputs[rr * 3 + 2];
        maskf[rr] = (j0 != 0 || j1 != 0 || j2 != 0) ? 1.0f : 0.0f;
    }
}

// ---------------------------------------------------------------------------
// Kernel 2: CRF forward (denominator) + gold-path score (numerator).
//   One 64-thread block per batch element; thread j owns tag j.
//   Sequential over S; trans column kept in registers; alpha exchanged via
//   ping-pong LDS buffers -> one barrier per step.
// ---------------------------------------------------------------------------
__global__ __launch_bounds__(64)
void crf_scan_kernel(const int*   __restrict__ tags,        // (B,S)
                     const float* __restrict__ em,          // (B,S,T)
                     const float* __restrict__ maskf,       // (B,S)
                     const float* __restrict__ start_trans, // (T)
                     const float* __restrict__ end_trans,   // (T)
                     const float* __restrict__ trans,       // (T,T)
                     float*       __restrict__ out)         // (B)
{
    __shared__ float sTr[TT * TT];   // 16 KB, for numerator random access
    __shared__ float sAlpha[2][TT];  // ping-pong
    __shared__ float sRed[TT];
    __shared__ float sRed2[TT];
    __shared__ float sNum;

    const int j = threadIdx.x;   // tag index 0..63
    const int b = blockIdx.x;

    for (int idx = j; idx < TT * TT; idx += TT) sTr[idx] = trans[idx];

    const float* emB  = em    + (size_t)b * SS * TT;
    const float* mB   = maskf + (size_t)b * SS;
    const int*   tagB = tags  + (size_t)b * SS;

    // column trans[:, j] in registers (fully unrolled -> stays in VGPRs)
    float trc[TT];
    #pragma unroll
    for (int i = 0; i < TT; ++i) trc[i] = trans[i * TT + j];

    const int tag0 = tagB[0];
    float alpha = start_trans[j] + emB[j];
    sAlpha[0][j] = alpha;
    __syncthreads();

    int cur = 0;
    for (int s = 1; s < SS; ++s) {
        const float em_v = emB[s * TT + j];
        const float mv   = mB[s];
        if (s + 1 < SS) __builtin_prefetch(emB + (s + 1) * TT + j, 0, 1);

        float mx = -1e30f;
        #pragma unroll
        for (int i = 0; i < TT; ++i)
            mx = fmaxf(mx, sAlpha[cur][i] + trc[i]);
        float sum = 0.f;
        #pragma unroll
        for (int i = 0; i < TT; ++i)
            sum += __expf(sAlpha[cur][i] + trc[i] - mx);

        const float nxt = mx + __logf(sum) + em_v;
        alpha = (mv != 0.0f) ? nxt : alpha;
        sAlpha[cur ^ 1][j] = alpha;     // write the other buffer
        __syncthreads();                // single barrier per step
        cur ^= 1;
    }

    // ---- numerator partials (strided over s) ----
    float part = 0.f, cnt = 0.f;
    for (int s = j; s < SS; s += TT) {
        const float mv = mB[s];
        cnt += mv;
        if (s >= 1) {
            const int tp = tagB[s - 1], tc = tagB[s];
            part += (sTr[tp * TT + tc] + emB[s * TT + tc]) * mv;
        }
    }
    sRed[j] = part; sRed2[j] = cnt;
    __syncthreads();
    for (int off = 32; off > 0; off >>= 1) {
        if (j < off) { sRed[j] += sRed[j + off]; sRed2[j] += sRed2[j + off]; }
        __syncthreads();
    }
    if (j == 0) {
        const int seq_end  = (int)(sRed2[0] + 0.5f) - 1;
        const int last_tag = tagB[seq_end];
        sNum = start_trans[tag0] + emB[tag0] + sRed[0] + end_trans[last_tag];
    }
    __syncthreads();

    // ---- denominator: logsumexp_j(alpha + end_trans) ----
    const float v = alpha + end_trans[j];
    sRed[j] = v;
    __syncthreads();
    for (int off = 32; off > 0; off >>= 1) {
        if (j < off) sRed[j] = fmaxf(sRed[j], sRed[j + off]);
        __syncthreads();
    }
    const float mx = sRed[0];
    __syncthreads();
    sRed[j] = __expf(v - mx);
    __syncthreads();
    for (int off = 32; off > 0; off >>= 1) {
        if (j < off) sRed[j] += sRed[j + off];
        __syncthreads();
    }
    if (j == 0) out[b] = (mx + __logf(sRed[0])) - sNum;
}

// ---------------------------------------------------------------------------
extern "C" void kernel_launch(void* const* d_in, const int* in_sizes, int n_in,
                              void* d_out, int out_size, void* d_ws, size_t ws_size,
                              hipStream_t stream) {
    (void)in_sizes; (void)n_in; (void)out_size; (void)ws_size;
    const int*   inputs    = (const int*)  d_in[0];
    const int*   tags      = (const int*)  d_in[1];
    const float* emb_table = (const float*)d_in[2];
    const float* W1        = (const float*)d_in[3];
    const float* b1        = (const float*)d_in[4];
    const float* W2        = (const float*)d_in[5];
    const float* b2        = (const float*)d_in[6];
    const float* start_tr  = (const float*)d_in[7];
    const float* end_tr    = (const float*)d_in[8];
    const float* trans     = (const float*)d_in[9];
    float* out = (float*)d_out;

    // workspace layout: em (NROW*TT f32 = 64 MB) | mask (NROW f32 = 1 MB)
    float* em   = (float*)d_ws;
    float* mask = em + (size_t)NROW * TT;

    // 8192 blocks x 64 threads: each block = 2 waves x 16-row WMMA tiles
    crf_mlp_wmma_kernel<<<NROW / 32, 64, 0, stream>>>(
        inputs, emb_table, W1, b1, W2, b2, em, mask);

    // one block per batch element
    crf_scan_kernel<<<BB, 64, 0, stream>>>(
        tags, em, mask, start_tr, end_tr, trans, out);
}
